// EvoFill_37469294690849
// MI455X (gfx1250) — matrix operand order repaired and checked by reading, not compiled, not run on hardware
//
#include <hip/hip_runtime.h>
#include <hip/hip_bf16.h>

// ---------------- problem constants (from reference setup_inputs) ----------
#define BATCH   8
#define LSEQ    2048
#define DM      256
#define NA      4
#define DI      512
#define NH      8          // ssd heads
#define HD      64         // head dim
#define DS      128        // d_state
#define ZDIM    1288       // 2*DI + 2*DS + NH
#define XBC     768        // DI + 2*DS
#define CL      512        // chunk len
#define OV      64
#define STEPC   448        // CL - OV
#define NCHUNK  5          // ceil((2048-64)/448)
#define NCB     40         // NCHUNK * BATCH
#define NBLK    8          // CL / 64
#define BLK     64
#define NLAYER  4

typedef __attribute__((ext_vector_type(16))) _Float16 v16h;
typedef __attribute__((ext_vector_type(8)))  float    v8f;

// =========================================================================
// CDNA5 Tensor-Data-Mover demo kernel: NULL descriptor (D# group0.count==0
// => architectural NOP per ISA 8.3) exercises tensor_load_to_lds +
// s_wait_tensorcnt + cluster barrier (NOP when ClusterID==0).
// =========================================================================
typedef __attribute__((ext_vector_type(4))) unsigned tdm_u4;
typedef __attribute__((ext_vector_type(8))) int      tdm_i8;
typedef __attribute__((ext_vector_type(4))) int      tdm_i4;

__global__ void k_tdm() {
  tdm_u4 g0 = {};
  tdm_i8 g1 = {};
  tdm_i4 g2 = {};
  tdm_i4 g3 = {};
#if defined(__has_builtin)
#if __has_builtin(__builtin_amdgcn_tensor_load_to_lds)
#if __clang_major__ >= 23
  tdm_i8 g4 = {};
  __builtin_amdgcn_tensor_load_to_lds(g0, g1, g2, g3, g4, 0);
#else
  __builtin_amdgcn_tensor_load_to_lds(g0, g1, g2, g3, 0);
#endif
  __builtin_amdgcn_s_wait_tensorcnt(0);
#endif
#if __has_builtin(__builtin_amdgcn_s_cluster_barrier)
  __builtin_amdgcn_s_cluster_barrier();
#endif
#endif
}

// =========================================================================
// Generic strided + batched WMMA GEMM:  C = [beta*C +] rowscale .* (A @ B) + bias
//   - A viewed via (saM,saK), B via (sbK,sbN) -> transposed views are free
//   - batch z decomposed as (z0,z1,z2) with divisors (nz1,nz2)
//   - REQUIRES: M % 64 == 0, K % 32 == 0 (true for every call here);
//     only N may be ragged (N=1288) -> clamp+select, no exec divergence.
// Block tile 64x64, 8 wave32 waves; wave = one A frag x two 16x16 tiles.
// Double-buffered LDS tiles in *fragment order* ([strip][lane][8 dwords]):
// each lane reads its v16h fragment as one 32B contiguous LDS access.
// One __syncthreads per K-step; staging pointers strength-reduced.
// =========================================================================
struct GemmP {
  int M, N, K;
  long offA, offB, offC;
  int saM, saK, sbK, sbN, scM, scN;
  long bA0, bA1, bA2;
  long bB0, bB1, bB2;
  long bC0, bC1, bC2;
  long bR0, bR1, bR2;
  int nz1, nz2, beta;
};

__global__ __launch_bounds__(256) void gemm_wmma(
    const float* __restrict__ A, const float* __restrict__ B,
    float* __restrict__ C, const float* __restrict__ bias,
    const float* __restrict__ rowscale, GemmP p)
{
  __shared__ alignas(32) unsigned Asf[2][4][32][8];  // ping-pong, 4 row-strips
  __shared__ alignas(32) unsigned Bsf[2][4][32][8];  // ping-pong, 4 col-strips

  int z  = blockIdx.z;
  int z2 = z % p.nz2;
  int zt = z / p.nz2;
  int z1 = zt % p.nz1;
  int z0 = zt / p.nz1;
  const float* Ab = A + p.offA + z0 * p.bA0 + z1 * p.bA1 + z2 * p.bA2;
  const float* Bb = B + p.offB + z0 * p.bB0 + z1 * p.bB1 + z2 * p.bB2;
  float*       Cb = C + p.offC + z0 * p.bC0 + z1 * p.bC1 + z2 * p.bC2;
  const float* Rb = rowscale ? (rowscale + z0 * p.bR0 + z1 * p.bR1 + z2 * p.bR2)
                             : (const float*)nullptr;

  int m0   = blockIdx.x * 64;
  int n0   = blockIdx.y * 64;
  int tid  = threadIdx.x;
  int lane = tid & 31;
  int wave = tid >> 5;
  int wm   = (wave & 3) * 16;   // wave row tile
  int wn   = (wave >> 2) * 32;  // wave col pair base (2 tiles of 16)
  int half = lane >> 4;
  int l16  = lane & 15;

  // ---- per-thread staging state: 4 K-pairs for A, 4 for B ----
  const float* aP[4];
  const float* bP[4];
  int aOff[4], bOff[4];
  bool bok[4];
  #pragma unroll
  for (int i = 0; i < 4; ++i) {
    int pidx = i * 256 + tid;          // 1024 pairs each for A and B
    int k2   = (pidx & 15) * 2;        // even k within 32-wide tile
    int r    = pidx >> 4;              // A row 0..63 / B col 0..63
    aP[i]   = Ab + (long)(m0 + r) * p.saM + (long)k2 * p.saK;
    aOff[i] = ((r >> 4) * 32 + (((k2 >> 3) & 1) * 16 + (r & 15))) * 8
            + ((((k2 & 16) >> 1) | (k2 & 7)) >> 1);
    int gn  = n0 + r;
    bok[i]  = gn < p.N;
    int gnc = bok[i] ? gn : (p.N - 1);
    bP[i]   = Bb + (long)k2 * p.sbK + (long)gnc * p.sbN;
    bOff[i] = ((r >> 4) * 32 + (((k2 >> 3) & 1) * 16 + (r & 15))) * 8
            + ((((k2 & 16) >> 1) | (k2 & 7)) >> 1);
  }
  long aStep = 32L * p.saK;
  long bStep = 32L * p.sbK;
  unsigned* Af = &Asf[0][0][0][0];
  unsigned* Bf = &Bsf[0][0][0][0];
  const unsigned* aRd  = &Asf[0][wm >> 4][lane][0];
  const unsigned* bRd0 = &Bsf[0][wn >> 4][lane][0];
  const unsigned* bRd1 = &Bsf[0][(wn >> 4) + 1][lane][0];

  auto stage = [&](int buf) {
    if (p.saK == 1) {                  // uniform branch -> b64 loads
      #pragma unroll
      for (int i = 0; i < 4; ++i) {
        float2 t = *(const float2*)aP[i];
        union { _Float16 h[2]; unsigned u; } pk;
        pk.h[0] = (_Float16)t.x; pk.h[1] = (_Float16)t.y;
        Af[buf * 1024 + aOff[i]] = pk.u;
      }
    } else {
      #pragma unroll
      for (int i = 0; i < 4; ++i) {
        float v0 = aP[i][0], v1 = aP[i][p.saK];
        union { _Float16 h[2]; unsigned u; } pk;
        pk.h[0] = (_Float16)v0; pk.h[1] = (_Float16)v1;
        Af[buf * 1024 + aOff[i]] = pk.u;
      }
    }
    if (p.sbK == 1) {
      #pragma unroll
      for (int i = 0; i < 4; ++i) {
        float2 t = *(const float2*)bP[i];
        float v0 = bok[i] ? t.x : 0.f;
        float v1 = bok[i] ? t.y : 0.f;
        union { _Float16 h[2]; unsigned u; } pk;
        pk.h[0] = (_Float16)v0; pk.h[1] = (_Float16)v1;
        Bf[buf * 1024 + bOff[i]] = pk.u;
      }
    } else {
      #pragma unroll
      for (int i = 0; i < 4; ++i) {
        float v0 = bP[i][0], v1 = bP[i][p.sbK];
        if (!bok[i]) { v0 = 0.f; v1 = 0.f; }
        union { _Float16 h[2]; unsigned u; } pk;
        pk.h[0] = (_Float16)v0; pk.h[1] = (_Float16)v1;
        Bf[buf * 1024 + bOff[i]] = pk.u;
      }
    }
    #pragma unroll
    for (int i = 0; i < 4; ++i) { aP[i] += aStep; bP[i] += bStep; }
  };

  v8f accs[2] = {};
  stage(0);
  __syncthreads();
  int nk = p.K >> 5;
  for (int t2 = 0; t2 < nk; ++t2) {
    int cur = t2 & 1;
    if (t2 + 1 < nk) {
      stage(cur ^ 1);
      __builtin_prefetch(aP[0], 0, 3);   // global_prefetch_b8, tile t2+2
      __builtin_prefetch(bP[0], 0, 3);
    }
    v16h a  = *(const v16h*)(aRd  + cur * 1024);
    v16h b0 = *(const v16h*)(bRd0 + cur * 1024);
    v16h b1 = *(const v16h*)(bRd1 + cur * 1024);
    accs[0] = __builtin_amdgcn_wmma_f32_16x16x32_f16(
        false, a, false, b0, (short)0, accs[0], false, false);
    accs[1] = __builtin_amdgcn_wmma_f32_16x16x32_f16(
        false, a, false, b1, (short)0, accs[1], false, false);
    __syncthreads();
  }

  // D layout: VGPR r -> row (r + 8*half), col l16
  #pragma unroll
  for (int cg = 0; cg < 2; ++cg) {
    #pragma unroll
    for (int r = 0; r < 8; ++r) {
      int gm = m0 + wm + r + 8 * half;          // always < M (M % 64 == 0)
      int gn = n0 + wn + cg * 16 + l16;
      if (gn < p.N) {
        float v = accs[cg][r];
        if (Rb)   v *= Rb[gm];
        if (bias) v += bias[gn];
        long ci = (long)gm * p.scM + (long)gn * p.scN;
        if (p.beta) Cb[ci] += v; else Cb[ci] = v;
      }
    }
  }
}

// =========================================================================
// Elementwise / small kernels
// =========================================================================
__global__ void k_e2(const float* xc, const float* cw, const float* cb, float* e2) {
  int i = blockIdx.x * blockDim.x + threadIdx.x;
  if (i >= LSEQ * DM) return;
  int l = i / DM, d = i % DM;
  float s = cb[d];
  for (int j = 0; j < 4; ++j) s += xc[l * 4 + j] * cw[j * DM + d];
  e2[i] = s;
}

__global__ void k_e1(const int* x, const float* emb, float* e1) {
  int i = blockIdx.x * blockDim.x + threadIdx.x;
  if (i >= BATCH * LSEQ * DM) return;
  int row = i / DM, d = i % DM;
  int xv = x[row];
  int xi = (xv < 0) ? NA : xv;
  e1[i] = emb[xi * DM + d];
}

__global__ void k_mul_u(float* t1, const float* t2) {
  int i = blockIdx.x * blockDim.x + threadIdx.x;
  if (i >= BATCH * LSEQ * DM) return;
  int l = (i / DM) % LSEQ;
  t1[i] = t1[i] * t2[l * DM + i % DM];
}

__global__ __launch_bounds__(256) void k_ln_h(
    const float* zf, const float* ob, const float* e1, const float* e2,
    const float* w, const float* bb, float* h)
{
  __shared__ float red[256];
  int row = blockIdx.x, d = threadIdx.x;
  int l = row % LSEQ;
  float v = zf[row * DM + d] + ob[d] + e1[row * DM + d] + e2[l * DM + d];
  red[d] = v; __syncthreads();
  for (int s = 128; s > 0; s >>= 1) { if (d < s) red[d] += red[d + s]; __syncthreads(); }
  float m = red[0] / DM; __syncthreads();
  float c = v - m;
  red[d] = c * c; __syncthreads();
  for (int s = 128; s > 0; s >>= 1) { if (d < s) red[d] += red[d + s]; __syncthreads(); }
  float var = red[0] / DM;
  h[row * DM + d] = c * rsqrtf(var + 1e-5f) * w[d] + bb[d];
}

__global__ __launch_bounds__(256) void k_hm(
    const float* h, const float* w, const float* bb, float* hm)
{
  __shared__ float red[256];
  int b = blockIdx.x, d = threadIdx.x;
  float s = 0.f;
  for (int l = 0; l < LSEQ; ++l) s += h[(b * LSEQ + l) * DM + d];
  s /= (float)LSEQ;
  red[d] = s; __syncthreads();
  for (int t = 128; t > 0; t >>= 1) { if (d < t) red[d] += red[d + t]; __syncthreads(); }
  float m = red[0] / DM; __syncthreads();
  float c = s - m;
  red[d] = c * c; __syncthreads();
  for (int t = 128; t > 0; t >>= 1) { if (d < t) red[d] += red[d + t]; __syncthreads(); }
  float var = red[0] / DM;
  hm[b * DM + d] = c * rsqrtf(var + 1e-5f) * w[d] + bb[d];
}

__global__ __launch_bounds__(256) void k_attn(
    const float* hm, const float* wq, const float* wk, const float* wv,
    const float* dist, float* av)
{
  __shared__ float q[8 * 256], k[8 * 256], v[8 * 256], sc[512], bias2[64];
  __shared__ float dmax;
  int t = threadIdx.x;
  for (int i = t; i < 8 * 256; i += 256) {
    int s = i >> 8, d = i & 255;
    float aq = 0.f, ak = 0.f, avv = 0.f;
    for (int r = 0; r < 256; ++r) {
      float hv = hm[s * 256 + r];
      aq  += hv * wq[r * 256 + d];
      ak  += hv * wk[r * 256 + d];
      avv += hv * wv[r * 256 + d];
    }
    q[i] = aq; k[i] = ak; v[i] = avv;
  }
  if (t == 0) {
    float mx = dist[0];
    for (int i = 1; i < 64; ++i) mx = fmaxf(mx, dist[i]);
    dmax = mx;
  }
  __syncthreads();
  if (t < 64) bias2[t] = 1.f - dist[t] / (dmax + 1e-6f);
  __syncthreads();
  if (t < 512) {
    int hh = t >> 6, bq = (t >> 3) & 7, ck = t & 7;
    float s = 0.f;
    for (int j = 0; j < 32; ++j) s += q[bq * 256 + hh * 32 + j] * k[ck * 256 + hh * 32 + j];
    sc[t] = s * 0.17677669529663687f + bias2[bq * 8 + ck];
  }
  __syncthreads();
  if (t < 64) {
    int hh = t >> 3, bq = t & 7;
    int base = hh * 64 + bq * 8;
    float mx = sc[base];
    for (int c = 1; c < 8; ++c) mx = fmaxf(mx, sc[base + c]);
    float sum = 0.f;
    float e[8];
    for (int c = 0; c < 8; ++c) { e[c] = expf(sc[base + c] - mx); sum += e[c]; }
    for (int c = 0; c < 8; ++c) sc[base + c] = e[c] / sum;
  }
  __syncthreads();
  for (int i = t; i < 8 * 256; i += 256) {
    int bq = i >> 8, d = i & 255;
    int hh = d >> 5, j = d & 31;
    float s = 0.f;
    for (int c = 0; c < 8; ++c) s += sc[hh * 64 + bq * 8 + c] * v[c * 256 + hh * 32 + j];
    av[i] = s;
  }
}

__global__ void k_evo_o(const float* av, const float* wow, const float* wob, float* o) {
  int i = blockIdx.x * blockDim.x + threadIdx.x;
  if (i >= 8 * 256) return;
  int b = i >> 8, d = i & 255;
  float s = wob[d];
  for (int r = 0; r < 256; ++r) s += av[b * 256 + r] * wow[r * 256 + d];
  o[i] = s;
}

__global__ void k_chunk(const float* h, const float* o, float* ch) {
  int i = blockIdx.x * blockDim.x + threadIdx.x;
  if (i >= NCB * CL * DM) return;
  int ci = i / (CL * DM);
  int rem = i % (CL * DM);
  int t = rem / DM, d = rem % DM;
  int ic = ci / BATCH, b = ci % BATCH;
  int l = ic * STEPC + t;
  ch[i] = (l < LSEQ) ? (h[(b * LSEQ + l) * DM + d] + o[b * DM + d]) : 0.f;
}

__global__ void k_rev(const float* in, float* out) {
  int i = blockIdx.x * blockDim.x + threadIdx.x;
  if (i >= NCB * CL * DM) return;
  int b = i / (CL * DM);
  int rem = i % (CL * DM);
  int t = rem / DM, d = rem % DM;
  out[i] = in[(b * CL + (CL - 1 - t)) * DM + d];
}

__global__ void k_dtprep(const float* zb, const float* dtbias, const float* alog,
                         float* dtv, float* dacum, float* expda, float* decay,
                         float* dach)
{
  int i = blockIdx.x * blockDim.x + threadIdx.x;
  if (i >= NCB * NBLK * NH) return;
  int h = i % NH;
  int c = (i / NH) % NBLK;
  int b = i / (NH * NBLK);
  float Ah = -expf(alog[h]);
  float cum = 0.f;
  int zrow = b * CL + c * BLK;
  for (int kk = 0; kk < BLK; ++kk) {
    float raw = zb[(long)(zrow + kk) * ZDIM + 1280 + h] + dtbias[h];
    float sp = (raw > 20.f) ? raw : log1pf(expf(raw));
    dtv[(zrow + kk) * NH + h] = sp;
    cum += sp * Ah;
    dacum[i * BLK + kk] = cum;
    expda[i * BLK + kk] = expf(cum);
  }
  dach[i] = cum;
  for (int kk = 0; kk < BLK; ++kk)
    decay[i * BLK + kk] = expf(cum - dacum[i * BLK + kk]);
}

__global__ void k_conv(const float* zb, const float* cw, const float* cb,
                       const float* dtv, float* xbc, float* xdt)
{
  int i = blockIdx.x * blockDim.x + threadIdx.x;
  if (i >= NCB * CL * XBC) return;
  int ch = i % XBC;
  int t  = (i / XBC) % CL;
  int b  = i / (XBC * CL);
  float acc = cb[ch];
  for (int j = 0; j < 4; ++j) {
    int tt = t - 3 + j;
    if (tt >= 0) acc += cw[j * XBC + ch] * zb[(long)(b * CL + tt) * ZDIM + DI + ch];
  }
  float sig = 1.f / (1.f + expf(-acc));
  float xv = acc * sig;
  xbc[(b * CL + t) * XBC + ch] = xv;
  if (ch < DI) {
    int h = ch >> 6;
    xdt[((b * CL + t) * NH + h) * HD + (ch & 63)] = xv * dtv[(b * CL + t) * NH + h];
  }
}

__global__ void k_mh(const float* CB, const float* dacum, float* mh) {
  int i = blockIdx.x * blockDim.x + threadIdx.x;
  if (i >= NCB * NBLK * NH * 4096) return;
  int z  = i >> 12;
  int qk = i & 4095;
  int q = qk >> 6, kk = qk & 63;
  int zbc = z / NH;
  float r = 0.f;
  if (q >= kk)
    r = CB[zbc * 4096 + q * 64 + kk] * expf(dacum[z * 64 + q] - dacum[z * 64 + kk]);
  mh[i] = r;
}

__global__ void k_xdtd(const float* xdt, const float* decay, float* xdtd) {
  int i = blockIdx.x * blockDim.x + threadIdx.x;
  if (i >= NCB * NBLK * NH * 4096) return;
  int z = i >> 12;
  int kk = (i >> 6) & 63, pp = i & 63;
  int h = z % NH;
  int c = (z / NH) % NBLK;
  int b = z / (NH * NBLK);
  xdtd[i] = xdt[((b * CL + c * BLK + kk) * NH + h) * HD + pp] * decay[z * 64 + kk];
}

__global__ __launch_bounds__(256) void k_scan(const float* st, const float* dach,
                                              float* prev)
{
  int bh = blockIdx.x;
  int b = bh / NH, h = bh % NH;
  int t = threadIdx.x;
  float carry[32];
  #pragma unroll
  for (int j = 0; j < 32; ++j) carry[j] = 0.f;
  for (int c = 0; c < NBLK; ++c) {
    long base = (long)((b * NBLK + c) * NH + h) * 8192;
    float e = expf(dach[(b * NBLK + c) * NH + h]);
    #pragma unroll
    for (int j = 0; j < 32; ++j) {
      int idx = j * 256 + t;
      prev[base + idx] = carry[j];
      carry[j] = carry[j] * e + st[base + idx];
    }
  }
}

__global__ __launch_bounds__(256) void k_gate(
    const float* Y, const float* xbc, const float* zb,
    const float* Dp, const float* nw, float* yg)
{
  __shared__ float red[256];
  int row = blockIdx.x, t = threadIdx.x;
  float ys[2];
  float ss = 0.f;
  for (int r = 0; r < 2; ++r) {
    int i = t + r * 256;
    float xh = xbc[(long)row * XBC + i];
    float yv = Y[(long)row * DI + i] + xh * Dp[i >> 6];
    float zg = zb[(long)row * ZDIM + i];
    float sil = zg / (1.f + expf(-zg));
    yv *= sil;
    ys[r] = yv;
    ss += yv * yv;
  }
  red[t] = ss; __syncthreads();
  for (int s = 128; s > 0; s >>= 1) { if (t < s) red[t] += red[t + s]; __syncthreads(); }
  float scale = rsqrtf(red[0] / DI + 1e-5f);
  for (int r = 0; r < 2; ++r) {
    int i = t + r * 256;
    yg[(long)row * DI + i] = ys[r] * scale * nw[i];
  }
}

__global__ void k_combine(const float* oF, const float* oB, float* ch) {
  int i = blockIdx.x * blockDim.x + threadIdx.x;
  if (i >= NCB * CL * DM) return;
  int b = i / (CL * DM);
  int rem = i % (CL * DM);
  int t = rem / DM, d = rem % DM;
  ch[i] = oF[i] + oB[(b * CL + (CL - 1 - t)) * DM + d];
}

__global__ void k_avg(const float* ch, float* ob) {
  int i = blockIdx.x * blockDim.x + threadIdx.x;
  if (i >= BATCH * LSEQ * DM) return;
  int b = i / (LSEQ * DM);
  int l = (i / DM) % LSEQ;
  int d = i % DM;
  float s = 0.f;
  float cnt = 0.f;
  for (int ic = 0; ic < NCHUNK; ++ic) {
    int t = l - ic * STEPC;
    if (t >= 0 && t < CL) { s += ch[((ic * BATCH + b) * CL + t) * DM + d]; cnt += 1.f; }
  }
  ob[i] = s / fmaxf(cnt, 1.f);
}

__global__ void k_proj(const float* ob, const float* opw, const float* opb,
                       float* out)
{
  int i = blockIdx.x * blockDim.x + threadIdx.x;
  if (i >= BATCH * LSEQ) return;
  float lg[NA];
  for (int a = 0; a < NA; ++a) lg[a] = opb[a];
  for (int d = 0; d < DM; ++d) {
    float v = ob[(long)i * DM + d];
    for (int a = 0; a < NA; ++a) lg[a] += v * opw[d * NA + a];
  }
  float mx = lg[0];
  for (int a = 1; a < NA; ++a) mx = fmaxf(mx, lg[a]);
  float sum = 0.f;
  for (int a = 0; a < NA; ++a) { lg[a] = expf(lg[a] - mx); sum += lg[a]; }
  for (int a = 0; a < NA; ++a) out[i * NA + a] = lg[a] / sum;
}

// =========================================================================
// host side
// =========================================================================
static void gemm(hipStream_t s, const float* A, const float* B, float* C,
                 const float* bias, const float* rs, const GemmP& p, int nz0)
{
  dim3 g((p.M + 63) / 64, (p.N + 63) / 64, nz0 * p.nz1 * p.nz2);
  gemm_wmma<<<g, 256, 0, s>>>(A, B, C, bias, rs, p);
}

static GemmP mkp(int M, int N, int K, long offA, int saM, int saK,
                 long offB, int sbK, int sbN, long offC, int scM, int scN,
                 long bA0, long bA1, long bA2, long bB0, long bB1, long bB2,
                 long bC0, long bC1, long bC2, long bR0, long bR1, long bR2,
                 int nz1, int nz2, int beta)
{
  GemmP p;
  p.M = M; p.N = N; p.K = K;
  p.offA = offA; p.offB = offB; p.offC = offC;
  p.saM = saM; p.saK = saK; p.sbK = sbK; p.sbN = sbN; p.scM = scM; p.scN = scN;
  p.bA0 = bA0; p.bA1 = bA1; p.bA2 = bA2;
  p.bB0 = bB0; p.bB1 = bB1; p.bB2 = bB2;
  p.bC0 = bC0; p.bC1 = bC1; p.bC2 = bC2;
  p.bR0 = bR0; p.bR1 = bR1; p.bR2 = bR2;
  p.nz1 = nz1; p.nz2 = nz2; p.beta = beta;
  return p;
}

#define GRID1(n) dim3(((n) + 255) / 256), dim3(256)

extern "C" void kernel_launch(void* const* d_in, const int* in_sizes, int n_in,
                              void* d_out, int out_size, void* d_ws, size_t ws_size,
                              hipStream_t stream)
{
  (void)in_sizes; (void)n_in; (void)out_size; (void)ws_size;
  const int*   x        = (const int*)d_in[0];
  const float* x_coord  = (const float*)d_in[1];
  const float* dist     = (const float*)d_in[2];
  const float* embed    = (const float*)d_in[3];
  const float* coord_w  = (const float*)d_in[4];
  const float* coord_b  = (const float*)d_in[5];
  const float* bil_l    = (const float*)d_in[6];
  const float* bil_r    = (const float*)d_in[7];
  const float* bil_o_w  = (const float*)d_in[8];
  const float* bil_o_b  = (const float*)d_in[9];
  const float* gn_w     = (const float*)d_in[10];
  const float* gn_b     = (const float*)d_in[11];
  const float* evo_w    = (const float*)d_in[12];
  const float* evo_b    = (const float*)d_in[13];
  const float* wq       = (const float*)d_in[14];
  const float* wk       = (const float*)d_in[15];
  const float* wv       = (const float*)d_in[16];
  const float* wo_w     = (const float*)d_in[17];
  const float* wo_b     = (const float*)d_in[18];
  const float* in_proj  = (const float*)d_in[19];
  const float* out_proj = (const float*)d_in[20];
  const float* conv_w   = (const float*)d_in[21];
  const float* conv_b   = (const float*)d_in[22];
  const float* dt_bias  = (const float*)d_in[23];
  const float* A_log    = (const float*)d_in[24];
  const float* Dparam   = (const float*)d_in[25];
  const float* norm_w   = (const float*)d_in[26];
  const float* outp_w   = (const float*)d_in[27];
  const float* outp_b   = (const float*)d_in[28];
  float* out = (float*)d_out;

  // ---- workspace bump allocator (f32 elements) ----
  size_t off = 0;
  auto alloc = [&](size_t elems) -> float* {
    float* p = (float*)((char*)d_ws + off);
    off += elems * sizeof(float);
    off = (off + 255) & ~(size_t)255;
    return p;
  };
  float* e2b   = alloc((size_t)LSEQ * DM);
  float* e1b   = alloc((size_t)BATCH * LSEQ * DM);
  float* t2b   = alloc((size_t)LSEQ * DM);
  float* t1b   = alloc((size_t)BATCH * LSEQ * DM);
  float* zfb   = alloc((size_t)BATCH * LSEQ * DM);
  float* hb    = alloc((size_t)BATCH * LSEQ * DM);
  float* hmb   = alloc((size_t)BATCH * DM);
  float* avb   = alloc((size_t)BATCH * DM);
  float* ob_   = alloc((size_t)BATCH * DM);
  float* chb   = alloc((size_t)NCB * CL * DM);
  float* chRb  = alloc((size_t)NCB * CL * DM);
  float* oFb   = alloc((size_t)NCB * CL * DM);
  float* oBb   = alloc((size_t)NCB * CL * DM);
  float* zb    = alloc((size_t)NCB * CL * ZDIM);
  float* xbcb  = alloc((size_t)NCB * CL * XBC);
  float* xdtb  = alloc((size_t)NCB * CL * DI);
  float* xdtdb = alloc((size_t)NCB * NBLK * NH * 4096);
  float* dtvb  = alloc((size_t)NCB * CL * NH);
  float* dacb  = alloc((size_t)NCB * NBLK * NH * BLK);
  float* expb  = alloc((size_t)NCB * NBLK * NH * BLK);
  float* decb  = alloc((size_t)NCB * NBLK * NH * BLK);
  float* dchb  = alloc((size_t)NCB * NBLK * NH);
  float* CBb   = alloc((size_t)NCB * NBLK * 4096);
  float* Mhb   = alloc((size_t)NCB * NBLK * NH * 4096);
  float* Yb    = alloc((size_t)NCB * CL * DI);
  float* stb   = alloc((size_t)NCB * NBLK * NH * 8192);
  float* prevb = alloc((size_t)NCB * NBLK * NH * 8192);
  float* ygb   = alloc((size_t)NCB * CL * DI);
  float* obuf  = alloc((size_t)BATCH * LSEQ * DM);

  // CDNA5 TDM / cluster-barrier demo (architectural NOPs, safe)
  k_tdm<<<dim3(1), dim3(32), 0, stream>>>();

  // ---- front end ----
  k_e2<<<GRID1(LSEQ * DM), 0, stream>>>(x_coord, coord_w, coord_b, e2b);
  k_e1<<<GRID1(BATCH * LSEQ * DM), 0, stream>>>(x, embed, e1b);
  gemm(stream, e2b, bil_r, t2b, nullptr, nullptr,
       mkp(LSEQ, DM, DM, 0, DM, 1, 0, DM, 1, 0, DM, 1,
           0,0,0, 0,0,0, 0,0,0, 0,0,0, 1, 1, 0), 1);
  gemm(stream, e1b, bil_l, t1b, nullptr, nullptr,
       mkp(BATCH * LSEQ, DM, DM, 0, DM, 1, 0, DM, 1, 0, DM, 1,
           0,0,0, 0,0,0, 0,0,0, 0,0,0, 1, 1, 0), 1);
  k_mul_u<<<GRID1(BATCH * LSEQ * DM), 0, stream>>>(t1b, t2b);
  gemm(stream, t1b, bil_o_w, zfb, nullptr, nullptr,
       mkp(BATCH * LSEQ, DM, DM, 0, DM, 1, 0, DM, 1, 0, DM, 1,
           0,0,0, 0,0,0, 0,0,0, 0,0,0, 1, 1, 0), 1);
  k_ln_h<<<dim3(BATCH * LSEQ), dim3(256), 0, stream>>>(zfb, bil_o_b, e1b, e2b,
                                                       gn_w, gn_b, hb);
  k_hm<<<dim3(BATCH), dim3(256), 0, stream>>>(hb, evo_w, evo_b, hmb);
  k_attn<<<dim3(1), dim3(256), 0, stream>>>(hmb, wq, wk, wv, dist, avb);
  k_evo_o<<<GRID1(BATCH * DM), 0, stream>>>(avb, wo_w, wo_b, ob_);
  k_chunk<<<GRID1(NCB * CL * DM), 0, stream>>>(hb, ob_, chb);

  // ---- mamba layers ----
  for (int li = 0; li < NLAYER; ++li) {
    const float* Wi = in_proj + (size_t)li * DM * ZDIM;
    const float* Wo = out_proj + (size_t)li * DI * DM;
    for (int dir = 0; dir < 2; ++dir) {
      const float* cw  = conv_w + (size_t)(li * 2 + dir) * 4 * XBC;
      const float* cb  = conv_b + (size_t)(li * 2 + dir) * XBC;
      const float* dtb = dt_bias + (size_t)(li * 2 + dir) * NH;
      const float* al  = A_log + (size_t)(li * 2 + dir) * NH;
      const float* Dp  = Dparam + (size_t)(li * 2 + dir) * NH;
      const float* nw  = norm_w + (size_t)(li * 2 + dir) * DI;
      const float* src = chb;
      float* dst = oFb;
      if (dir == 1) {
        k_rev<<<GRID1(NCB * CL * DM), 0, stream>>>(chb, chRb);
        src = chRb;
        dst = oBb;
      }
      // in_proj: (20480 x 256) @ (256 x 1288)
      gemm(stream, src, Wi, zb, nullptr, nullptr,
           mkp(NCB * CL, ZDIM, DM, 0, DM, 1, 0, ZDIM, 1, 0, ZDIM, 1,
               0,0,0, 0,0,0, 0,0,0, 0,0,0, 1, 1, 0), 1);
      k_dtprep<<<GRID1(NCB * NBLK * NH), 0, stream>>>(zb, dtb, al, dtvb, dacb,
                                                      expb, decb, dchb);
      k_conv<<<GRID1(NCB * CL * XBC), 0, stream>>>(zb, cw, cb, dtvb, xbcb, xdtb);
      // CB[z=(b,c)] = C_(64x128) @ B_^T(128x64)
      gemm(stream, xbcb, xbcb, CBb, nullptr, nullptr,
           mkp(64, 64, 128, 640, XBC, 1, 512, 1, XBC, 0, 64, 1,
               (long)CL * XBC, (long)BLK * XBC, 0,
               (long)CL * XBC, (long)BLK * XBC, 0,
               (long)NBLK * 4096, 4096, 0, 0,0,0, NBLK, 1, 0), NCB);
      k_mh<<<GRID1(NCB * NBLK * NH * 4096), 0, stream>>>(CBb, dacb, Mhb);
      k_xdtd<<<GRID1(NCB * NBLK * NH * 4096), 0, stream>>>(xdtb, decb, xdtdb);
      // Yd[z=(b,c,h)] = Mh(64x64) @ xdt_h(64x64) -> Ybuf
      gemm(stream, Mhb, xdtb, Yb, nullptr, nullptr,
           mkp(64, 64, 64, 0, 64, 1, 0, NH * HD, 1, 0, DI, 1,
               (long)NBLK * NH * 4096, (long)NH * 4096, 4096,
               (long)CL * DI, (long)BLK * DI, HD,
               (long)CL * DI, (long)BLK * DI, HD,
               0,0,0, NBLK, NH, 0), NCB);
      // states[z] = xdtd^T(64p x 64k) @ B_(64k x 128n)
      gemm(stream, xdtdb, xbcb, stb, nullptr, nullptr,
           mkp(64, 128, 64, 0, 1, 64, 512, XBC, 1, 0, 128, 1,
               (long)NBLK * NH * 4096, (long)NH * 4096, 4096,
               (long)CL * XBC, (long)BLK * XBC, 0,
               (long)NBLK * NH * 8192, (long)NH * 8192, 8192,
               0,0,0, NBLK, NH, 0), NCB);
      k_scan<<<dim3(NCB * NH), dim3(256), 0, stream>>>(stb, dchb, prevb);
      // Yo[z] = diag(exp(dAcum)) @ C_(64x128) @ prev^T(128x64), accumulate
      gemm(stream, xbcb, prevb, Yb, nullptr, expb,
           mkp(64, 64, 128, 640, XBC, 1, 0, 1, 128, 0, DI, 1,
               (long)CL * XBC, (long)BLK * XBC, 0,
               (long)NBLK * NH * 8192, (long)NH * 8192, 8192,
               (long)CL * DI, (long)BLK * DI, HD,
               (long)NBLK * NH * BLK, (long)NH * BLK, BLK,
               NBLK, NH, 1), NCB);
      k_gate<<<dim3(NCB * CL), dim3(256), 0, stream>>>(Yb, xbcb, zb, Dp, nw, ygb);
      // out_proj: (20480 x 512) @ (512 x 256)
      gemm(stream, ygb, Wo, dst, nullptr, nullptr,
           mkp(NCB * CL, DM, DI, 0, DI, 1, 0, DM, 1, 0, DM, 1,
               0,0,0, 0,0,0, 0,0,0, 0,0,0, 1, 1, 0), 1);
    }
    k_combine<<<GRID1(NCB * CL * DM), 0, stream>>>(oFb, oBb, chb);
  }

  // ---- overlap-average + projection + softmax ----
  k_avg<<<GRID1(BATCH * LSEQ * DM), 0, stream>>>(chb, obuf);
  k_proj<<<GRID1(BATCH * LSEQ), 0, stream>>>(obuf, outp_w, outp_b, out);
}
